// MultiHeadedDotAttention_48842368090575
// MI455X (gfx1250) — compile-verified, matrix operand order checked by reading, not run on hardware
//
#include <hip/hip_runtime.h>

// ---------------------------------------------------------------------------
// Relation-aware multi-head attention for gfx1250 (MI455X).
// bf16 WMMA everywhere; double-buffered TDM (tensor_load_to_lds) GEMM tiles.
// ---------------------------------------------------------------------------

typedef __bf16        v16bf __attribute__((ext_vector_type(16)));
typedef float         v8f   __attribute__((ext_vector_type(8)));
typedef unsigned int  v4u   __attribute__((ext_vector_type(4)));
typedef int           v4i   __attribute__((ext_vector_type(4)));
typedef int           v8i   __attribute__((ext_vector_type(8)));

#define BB 16       // batch
#define LL 512      // sequence length
#define DD 1024     // model dim
#define HD 8        // heads
#define DK 128      // head dim
#define MM (BB*LL)  // GEMM M = 8192

#if __has_builtin(__builtin_amdgcn_tensor_load_to_lds)
#define USE_TDM 1
#else
#define USE_TDM 0
#endif

__device__ __forceinline__ unsigned short f2bf(float f) {
  unsigned int u = __float_as_uint(f);
  u += 0x7fffu + ((u >> 16) & 1u);          // round-to-nearest-even
  return (unsigned short)(u >> 16);
}

__device__ __forceinline__ v8f wmma_bf16(v16bf a, v16bf b, v8f c) {
  return __builtin_amdgcn_wmma_f32_16x16x32_bf16(false, a, false, b, (short)0, c,
                                                 false, false);
}

// same-wave LDS store->load ordering fence (DS ops are in-order per wave;
// this blocks compiler reordering and makes the dependency explicit)
__device__ __forceinline__ void lds_fence() {
  asm volatile("s_wait_dscnt 0x0" ::: "memory");
}

#if USE_TDM
// TDM: DMA a rows x 32-col bf16 tile (row stride 1024 elements) into LDS.
// D# per CDNA5 ISA §8: group0 = {count/type/lds_addr/global_addr},
// group1 = {data_size, tensor dims, tile dims, dim0 stride}.
__device__ __forceinline__ void tdm_load_tile(
    const unsigned short* g, unsigned int lds_off, int rows) {
  unsigned long long ga = (unsigned long long)(uintptr_t)g;
  v4u g0 = { 1u,                                   // count = 1 (valid D#)
             lds_off,                              // lds_addr (bytes)
             (unsigned int)ga,                     // global_addr[31:0]
             ((unsigned int)(ga >> 32) & 0x01ffffffu) | (2u << 30) }; // type=2
  v8i g1 = { (int)(1u << 16),                      // data_size = 1 -> 2 bytes
             (int)(1024u << 16),                   // tensor_dim0 = 1024 (lo16)
             (int)((1024u & 0xffffu) << 16),       // dim0 hi=0 | tensor_dim1 lo
             (int)(32u << 16),                     // dim1 hi=0 | tile_dim0 = 32
             rows,                                 // tile_dim1 = rows
             (int)1024,                            // tensor_dim0_stride = 1024
             0, 0 };
  v4i z = { 0, 0, 0, 0 };
#if __clang_major__ >= 23
  v8i z8 = { 0, 0, 0, 0, 0, 0, 0, 0 };
  __builtin_amdgcn_tensor_load_to_lds(g0, g1, z, z, z8, 0);
#else
  __builtin_amdgcn_tensor_load_to_lds(g0, g1, z, z, 0);
#endif
}

__device__ __forceinline__ void tdm_wait() {
#if __has_builtin(__builtin_amdgcn_s_wait_tensorcnt)
  __builtin_amdgcn_s_wait_tensorcnt(0);
#else
  asm volatile("s_wait_tensorcnt 0x0" ::: "memory");
#endif
  asm volatile("" ::: "memory");
}
#endif  // USE_TDM

// ---------------------------------------------------------------------------
// Kernel 0a: 8 weight matrices f32 [D][D] -> bf16 transposed [N][K]
// order: Wq, Wk0..2, Wv0..2, Wo
// ---------------------------------------------------------------------------
__global__ __launch_bounds__(256) void prep_weights(
    const float* __restrict__ Wq, const float* __restrict__ Wk,
    const float* __restrict__ Wv, const float* __restrict__ Wo,
    unsigned short* __restrict__ WT) {
  size_t idx = (size_t)blockIdx.x * 256 + threadIdx.x;   // over 8*D*D
  int mat = (int)(idx / ((size_t)DD * DD));
  size_t rem = idx % ((size_t)DD * DD);
  int n = (int)(rem / DD);
  int k = (int)(rem % DD);
  const float* src = (mat == 0) ? Wq
                   : (mat <= 3) ? Wk + (size_t)(mat - 1) * DD * DD
                   : (mat <= 6) ? Wv + (size_t)(mat - 4) * DD * DD
                                : Wo;
  WT[idx] = f2bf(src[(size_t)k * DD + n]);
}

// ---------------------------------------------------------------------------
// Kernel 0b: convert query/key/value f32 [M,D] -> bf16 once (float4 granules)
// ---------------------------------------------------------------------------
__global__ __launch_bounds__(256) void convert_x(
    const float* __restrict__ q, const float* __restrict__ k,
    const float* __restrict__ v, unsigned short* __restrict__ Xb) {
  const size_t per = (size_t)MM * DD;
  size_t base = ((size_t)blockIdx.x * 256 + threadIdx.x) * 4;  // over 3*per
  int mat = (int)(base / per);
  size_t rem = base % per;
  const float* src = (mat == 0) ? q : (mat == 1) ? k : v;
  float4 val = *(const float4*)(src + rem);
  unsigned short* dst = Xb + (size_t)mat * per + rem;
  dst[0] = f2bf(val.x);
  dst[1] = f2bf(val.y);
  dst[2] = f2bf(val.z);
  dst[3] = f2bf(val.w);
}

// ---------------------------------------------------------------------------
// Kernel 1/3: bf16 WMMA GEMM  C[M,N] = A[M,K](bf16) * BT[N,K](bf16) + bias
// M=8192, N=K=1024.  Block tile 128x256, 8 waves (2x4), wave tile 64x64
// (16 WMMAs per wave per k-step).  LDS tiles double-buffered and staged by
// the Tensor Data Mover: DMA of tile k+1 overlaps WMMA of tile k.
// mode 0: f32 row-major [M,N]        (final output)
// mode 1: bf16 [B,H,L,dk]            (Q, K projections)
// mode 2: bf16 [B,H,dk,L] transposed (V projections)
// ---------------------------------------------------------------------------
#define Bb_M 128
#define Bb_N 256
#define Bb_K 32
#define NKS (DD / Bb_K)   // 32 k-steps

__global__ __launch_bounds__(256) void gemm_wmma(
    const unsigned short* __restrict__ A, const unsigned short* __restrict__ BT,
    const float* __restrict__ bias, void* __restrict__ Out, int mode) {
  __shared__ unsigned short As[2][Bb_M][Bb_K];   // 2 x 8 KB
  __shared__ unsigned short Bs[2][Bb_N][Bb_K];   // 2 x 16 KB

  const int tid  = threadIdx.x;
  const int lane = tid & 31;
  const int wave = tid >> 5;
  const int l16  = lane & 15;
  const int lh   = lane >> 4;
  const int wm   = wave >> 2;   // 0..1  (64-row slice)
  const int wn   = wave & 3;    // 0..3  (64-col slice)

  const int m0 = blockIdx.x * Bb_M;
  const int n0 = blockIdx.y * Bb_N;

#if USE_TDM
  const unsigned int ldsA[2] = { (unsigned int)(uintptr_t)&As[0][0][0],
                                 (unsigned int)(uintptr_t)&As[1][0][0] };
  const unsigned int ldsB[2] = { (unsigned int)(uintptr_t)&Bs[0][0][0],
                                 (unsigned int)(uintptr_t)&Bs[1][0][0] };
#endif

  // stage tile `ks` into buffer `buf`
  auto stage = [&](int ks, int buf) {
#if USE_TDM
    if (wave == 0) {
      tdm_load_tile(A  + (size_t)m0 * DD + ks * Bb_K, ldsA[buf], Bb_M);
      tdm_load_tile(BT + (size_t)n0 * DD + ks * Bb_K, ldsB[buf], Bb_N);
    }
#else
    const int k0 = ks * Bb_K;
#pragma unroll
    for (int i = 0; i < 4; ++i) {                  // A: 1024 8-byte units
      int u = tid + 256 * i;
      int r = u >> 3, c4 = u & 7;
      *(unsigned long long*)&As[buf][r][c4 * 4] =
          ((const unsigned long long*)(A + (size_t)(m0 + r) * DD + k0))[c4];
    }
#pragma unroll
    for (int i = 0; i < 8; ++i) {                  // B: 2048 8-byte units
      int u = tid + 256 * i;
      int r = u >> 3, c4 = u & 7;
      *(unsigned long long*)&Bs[buf][r][c4 * 4] =
          ((const unsigned long long*)(BT + (size_t)(n0 + r) * DD + k0))[c4];
    }
#endif
  };

  v8f acc[4][4] = {};

  // prologue: fill buffer 0
  stage(0, 0);
#if USE_TDM
  if (wave == 0) tdm_wait();
#endif
  __syncthreads();

  for (int ks = 0; ks < NKS; ++ks) {
    const int cur = ks & 1;
    if (ks + 1 < NKS) stage(ks + 1, cur ^ 1);      // DMA overlaps compute

    v16bf af[4];
#pragma unroll
    for (int mt = 0; mt < 4; ++mt)
      af[mt] = *(const v16bf*)&As[cur][wm * 64 + mt * 16 + l16][lh * 16];
#pragma unroll
    for (int nt = 0; nt < 4; ++nt) {
      v16bf bf = *(const v16bf*)&Bs[cur][wn * 64 + nt * 16 + l16][lh * 16];
#pragma unroll
      for (int mt = 0; mt < 4; ++mt)
        acc[mt][nt] = wmma_bf16(af[mt], bf, acc[mt][nt]);
    }

#if USE_TDM
    if (wave == 0 && ks + 1 < NKS) tdm_wait();     // next tile landed
#endif
    __syncthreads();   // publishes buf cur^1, recycles buf cur
  }

  // ---- epilogue ----
#pragma unroll
  for (int mt = 0; mt < 4; ++mt)
#pragma unroll
    for (int nt = 0; nt < 4; ++nt)
#pragma unroll
      for (int e = 0; e < 8; ++e) {
        int m = m0 + wm * 64 + mt * 16 + e + lh * 8;
        int n = n0 + wn * 64 + nt * 16 + l16;
        float val = acc[mt][nt][e] + bias[n];
        if (mode == 0) {
          ((float*)Out)[(size_t)m * DD + n] = val;
        } else {
          int b = m / LL, l = m % LL, h = n / DK, d = n % DK;
          size_t off = (mode == 1)
              ? ((((size_t)b * HD + h) * LL + l) * DK + d)     // [B,H,L,dk]
              : ((((size_t)b * HD + h) * DK + d) * LL + l);    // [B,H,dk,L]
          ((unsigned short*)Out)[off] = f2bf(val);
        }
      }
}

// ---------------------------------------------------------------------------
// Kernel 2: relation-selected flash attention (one 16-row q-tile per wave).
// ---------------------------------------------------------------------------
__global__ __launch_bounds__(128) void attn_kernel(
    const unsigned short* __restrict__ Qb,  // [B,H,L,dk] bf16
    const unsigned short* __restrict__ Kb,  // [3,B,H,L,dk] bf16
    const unsigned short* __restrict__ Vt,  // [3,B,H,dk,L] bf16
    const int* __restrict__ flag,           // [B,L,L]
    const unsigned char* __restrict__ mask, // [B,L]
    unsigned short* __restrict__ O) {       // [B,L,H,dk] bf16
  __shared__ unsigned short Pls[4][3][16 * 32];   // per-wave private regions

  const int lane = threadIdx.x & 31;
  const int wave = threadIdx.x >> 5;
  const int l16  = lane & 15;
  const int lh   = lane >> 4;

  const int nqt = LL / 64;
  const int bh  = blockIdx.x / nqt;
  const int qt  = blockIdx.x % nqt;
  const int b = bh / HD, h = bh % HD;
  const int q0 = qt * 64 + wave * 16;

  const float scale = 0.08838834764831845f;   // 1/sqrt(128)

  v16bf qf[4];
  const unsigned short* qbase =
      Qb + (((size_t)b * HD + h) * LL + q0 + l16) * DK;
#pragma unroll
  for (int kk = 0; kk < 4; ++kk)
    qf[kk] = *(const v16bf*)(qbase + kk * 32 + lh * 16);

  float mrow[8], lrow[8];
  v8f acc[8] = {};
#pragma unroll
  for (int e = 0; e < 8; ++e) { mrow[e] = -1e30f; lrow[e] = 0.0f; }

  for (int kb = 0; kb < LL / 32; ++kb) {
    const int c0 = kb * 32;
    if (kb + 1 < LL / 32)
      __builtin_prefetch(&flag[((size_t)b * LL + q0) * LL + c0 + 32], 0, 0);

    // ---- scores for 3 relation branches: S_r = Q * K_r^T ----
    v8f sacc[3][2] = {};
#pragma unroll
    for (int r = 0; r < 3; ++r) {
      const unsigned short* kbase =
          Kb + ((((size_t)r * BB + b) * HD + h) * LL) * DK;
#pragma unroll
      for (int ct = 0; ct < 2; ++ct) {
        const unsigned short* kr = kbase + (size_t)(c0 + ct * 16 + l16) * DK;
#pragma unroll
        for (int kk = 0; kk < 4; ++kk) {
          v16bf bf = *(const v16bf*)(kr + kk * 32 + lh * 16);
          sacc[r][ct] = wmma_bf16(qf[kk], bf, sacc[r][ct]);
        }
      }
    }

    // ---- per-element relation select + mask ----
    float smer[2][8];
    int   fl[2][8];
#pragma unroll
    for (int ct = 0; ct < 2; ++ct) {
      const int col = c0 + ct * 16 + l16;
      const bool mk = mask[(size_t)b * LL + col] != 0;
#pragma unroll
      for (int e = 0; e < 8; ++e) {
        const int row = q0 + e + lh * 8;
        int f = flag[((size_t)b * LL + row) * LL + col];
        fl[ct][e] = f;
        float s = (f == 0) ? sacc[0][ct][e]
                : (f == 1) ? sacc[1][ct][e]
                           : sacc[2][ct][e];
        s *= scale;
        smer[ct][e] = mk ? s : -1e9f;
      }
    }

    // ---- online softmax (16-lane row reductions) ----
    float alpha[8], mnew[8];
#pragma unroll
    for (int e = 0; e < 8; ++e) {
      float v = fmaxf(smer[0][e], smer[1][e]);
#pragma unroll
      for (int off = 8; off > 0; off >>= 1)
        v = fmaxf(v, __shfl_xor(v, off, 32));
      mnew[e]  = fmaxf(mrow[e], v);
      alpha[e] = __expf(mrow[e] - mnew[e]);
      mrow[e]  = mnew[e];
    }
    float p[2][8];
#pragma unroll
    for (int e = 0; e < 8; ++e) {
      p[0][e] = __expf(smer[0][e] - mnew[e]);
      p[1][e] = __expf(smer[1][e] - mnew[e]);
      float rs = p[0][e] + p[1][e];
#pragma unroll
      for (int off = 8; off > 0; off >>= 1)
        rs += __shfl_xor(rs, off, 32);
      lrow[e] = lrow[e] * alpha[e] + rs;
    }
#pragma unroll
    for (int nt = 0; nt < 8; ++nt)
#pragma unroll
      for (int e = 0; e < 8; ++e)
        acc[nt][e] *= alpha[e];

    // ---- relation-split P -> LDS (C-layout -> A-layout transpose) ----
    unsigned short* pw = &Pls[wave][0][0];
#pragma unroll
    for (int r = 0; r < 3; ++r)
#pragma unroll
      for (int ct = 0; ct < 2; ++ct)
#pragma unroll
        for (int e = 0; e < 8; ++e) {
          float pv = (fl[ct][e] == r) ? p[ct][e] : 0.0f;
          int row = e + lh * 8;
          pw[r * 512 + row * 32 + ct * 16 + l16] = f2bf(pv);
        }
    lds_fence();   // per-wave region; same-wave DS in-order, fence compiler

    // ---- P_r (16x32) * V_r (32x128) ----
#pragma unroll
    for (int r = 0; r < 3; ++r) {
      v16bf pf = *(const v16bf*)&Pls[wave][r][l16 * 32 + lh * 16];
      const unsigned short* vb =
          Vt + ((((size_t)r * BB + b) * HD + h) * DK) * LL;
#pragma unroll
      for (int nt = 0; nt < 8; ++nt) {
        v16bf bf = *(const v16bf*)(vb + (size_t)(nt * 16 + l16) * LL +
                                   c0 + lh * 16);
        acc[nt] = wmma_bf16(pf, bf, acc[nt]);
      }
    }
  }

  // ---- epilogue: normalize and store bf16 [B,L,H,dk] ----
#pragma unroll
  for (int e = 0; e < 8; ++e) {
    const float inv = 1.0f / lrow[e];
    const int row = q0 + e + lh * 8;
#pragma unroll
    for (int nt = 0; nt < 8; ++nt)
      O[(((size_t)b * LL + row) * HD + h) * DK + nt * 16 + l16] =
          f2bf(acc[nt][e] * inv);
  }
}

// ---------------------------------------------------------------------------
// Host-side launcher
// ---------------------------------------------------------------------------
extern "C" void kernel_launch(void* const* d_in, const int* in_sizes, int n_in,
                              void* d_out, int out_size, void* d_ws,
                              size_t ws_size, hipStream_t stream) {
  const float* query = (const float*)d_in[0];
  const float* key_  = (const float*)d_in[1];
  const float* value = (const float*)d_in[2];
  const int*   flag  = (const int*)d_in[3];
  const unsigned char* mask = (const unsigned char*)d_in[4];
  const float* Wq = (const float*)d_in[5];
  const float* bq = (const float*)d_in[6];
  const float* Wk = (const float*)d_in[7];
  const float* bk = (const float*)d_in[8];
  const float* Wv = (const float*)d_in[9];
  const float* bv = (const float*)d_in[10];
  const float* Wo = (const float*)d_in[11];
  const float* bo = (const float*)d_in[12];

  // workspace layout (bf16 element counts); per == qn == 8388608
  const size_t per = (size_t)MM * DD;
  const size_t qn  = (size_t)BB * HD * LL * DK;
  unsigned short* WT = (unsigned short*)d_ws;      // 8*D*D
  unsigned short* Xb = WT + (size_t)8 * DD * DD;   // 3*per (q,k,v bf16)
  unsigned short* Qb = Xb + 3 * per;
  unsigned short* Kb = Qb + qn;
  unsigned short* Vb = Kb + 3 * qn;
  unsigned short* Ob = Vb + 3 * qn;                // per (attn out bf16)

  // 0) one-time conversions
  prep_weights<<<(8 * DD * DD) / 256, 256, 0, stream>>>(Wq, Wk, Wv, Wo, WT);
  convert_x<<<(unsigned)(3 * per / 4 / 256), 256, 0, stream>>>(query, key_,
                                                               value, Xb);

  // 1) projections (all-bf16 WMMA GEMMs, double-buffered TDM tiles)
  dim3 gg(MM / Bb_M, DD / Bb_N);
  gemm_wmma<<<gg, 256, 0, stream>>>(Xb, WT, bq, Qb, 1);
  for (int r = 0; r < 3; ++r)
    gemm_wmma<<<gg, 256, 0, stream>>>(Xb + per, WT + (size_t)(1 + r) * DD * DD,
                                      bk + (size_t)r * DD, Kb + (size_t)r * qn, 1);
  for (int r = 0; r < 3; ++r)
    gemm_wmma<<<gg, 256, 0, stream>>>(Xb + 2 * per,
                                      WT + (size_t)(4 + r) * DD * DD,
                                      bv + (size_t)r * DD, Vb + (size_t)r * qn, 2);

  // 2) relation-selected flash attention
  attn_kernel<<<BB * HD * (LL / 64), 128, 0, stream>>>(Qb, Kb, Vb, flag, mask,
                                                       Ob);

  // 3) output projection -> f32 [B,L,D]
  gemm_wmma<<<gg, 256, 0, stream>>>(Ob, WT + (size_t)7 * DD * DD, bo, d_out, 0);

  (void)in_sizes; (void)n_in; (void)out_size; (void)ws_size;
}
